// CELPNet_87771951661269
// MI455X (gfx1250) — matrix operand order repaired and checked by reading, not compiled, not run on hardware
//
#include <hip/hip_runtime.h>

typedef __attribute__((ext_vector_type(16))) __bf16 v16bf;
typedef __attribute__((ext_vector_type(8)))  float  v8f;

// ---------- bf16 helpers (raw u16 storage) ----------
__device__ __forceinline__ unsigned short f2bf(float f) {
    union { float f; unsigned u; } v; v.f = f;
    unsigned u = v.u;
    unsigned r = (u + 0x7FFFu + ((u >> 16) & 1u)) >> 16;
    return (unsigned short)r;
}
__device__ __forceinline__ float bf2f(unsigned short b) {
    union { unsigned u; float f; } v; v.u = ((unsigned)b) << 16;
    return v.f;
}

// ---------- A-fragment loader: 16x32 bf16, contiguous k = 16*g + s (same as pack_w) ----------
// lane L: row = L%16, g = L/16 ; vgpr j holds K = g*16 + j*2 + {0,1}
__device__ __forceinline__ v16bf load_afrag(const unsigned short* __restrict__ ak, int g) {
    union { v16bf v; unsigned u[8]; } af;
#pragma unroll
    for (int j = 0; j < 8; ++j) {
        int k = (g << 4) + (j << 1);
        af.u[j] = *reinterpret_cast<const unsigned*>(ak + k);
    }
    return af.v;
}

__device__ __forceinline__ v8f wmma_bf16(v16bf a, v16bf b, v8f c) {
    return __builtin_amdgcn_wmma_f32_16x16x32_bf16(false, a, false, b, (short)0, c, false, false);
}

// ---------- weight packer: W[k][n] -> B-fragment order ----------
// dst[((kt*ntiles+nt)*32 + lane)*16 + e] = W[kt*32 + (lane/16)*16 + e][nt*16 + lane%16]
// mode 0: plain fp32 [Ksrc, Nsrc] row-major (zero padded past Ksrc/Nsrc)
// mode 1: conv weight [O=256, I=256, 3]; W[k][n] = w[n, k%256, k/256]
// mode 2: output head: n<40 -> sdo_w[k*40+n]; n==40 -> gain_w[k]; else 0
__global__ void pack_w(const float* __restrict__ W0, const float* __restrict__ W1,
                       int mode, int Ksrc, int Nsrc, int ntiles,
                       unsigned short* __restrict__ dst) {
    int lane = threadIdx.x;
    int kt = blockIdx.x / ntiles, nt = blockIdx.x % ntiles;
    size_t base = ((size_t)blockIdx.x * 32 + lane) * 16;
#pragma unroll
    for (int e = 0; e < 16; ++e) {
        int k = (kt << 5) + ((lane >> 4) << 4) + e;
        int n = (nt << 4) + (lane & 15);
        float v = 0.f;
        if (mode == 0) {
            if (k < Ksrc && n < Nsrc) v = W0[(size_t)k * Nsrc + n];
        } else if (mode == 1) {
            int i = k & 255, kk = k >> 8;
            v = W0[((size_t)n * 256 + i) * 3 + kk];
        } else {
            if (k < Ksrc) {
                if (n < 40) v = W0[(size_t)k * 40 + n];
                else if (n == 40) v = W1[k];
            }
        }
        dst[base + e] = f2bf(v);
    }
}

// ---------- generic bf16 WMMA GEMM: O = act(A @ W + bias), O bf16 [M,N] ----------
// A row r -> element offset (r/rpg)*gstride + (r%rpg)*rstride  (overlapping rows for conv)
__global__ __launch_bounds__(128) void gemm_bf16(
    const unsigned short* __restrict__ A, int rpg, long gstride, int rstride,
    const unsigned short* __restrict__ Bp, int ktiles, int ntilesB,
    const float* __restrict__ bias, int act,
    unsigned short* __restrict__ O, int N) {
    int wave = threadIdx.x >> 5, lane = threadIdx.x & 31;
    int nt = blockIdx.y * 4 + wave;
    int r0 = blockIdx.x << 4;
    int m = lane & 15, g = lane >> 4;
    int row = r0 + m;
    long arow = (long)(row / rpg) * gstride + (long)(row % rpg) * rstride;
    const unsigned short* Ab = A + arow;
    v8f acc;
#pragma unroll
    for (int i = 0; i < 8; ++i) acc[i] = 0.f;
    for (int kt = 0; kt < ktiles; ++kt) {
        v16bf a = load_afrag(Ab + (kt << 5), g);
        v16bf b = *reinterpret_cast<const v16bf*>(Bp + ((size_t)(kt * ntilesB + nt) * 32 + lane) * 16);
        acc = wmma_bf16(a, b, acc);
    }
    int n = (nt << 4) + m;
    float bv = bias ? bias[n] : 0.f;
#pragma unroll
    for (int v = 0; v < 8; ++v) {
        int orow = r0 + v + (g << 3);
        float x = acc[v] + bv;
        if (act == 1) x = tanhf(x);
        O[(size_t)orow * N + n] = f2bf(x);
    }
}

// ---------- fused GRU step: h' = (1-z)*n + z*h ; 48 WMMAs per wave ----------
__global__ __launch_bounds__(128) void gru_step(
    const unsigned short* __restrict__ xb, const unsigned short* __restrict__ hb,
    const float* __restrict__ hf,
    const unsigned short* __restrict__ Pwi, const unsigned short* __restrict__ Pwh,
    const float* __restrict__ bi, const float* __restrict__ bh,
    float* __restrict__ hfo, unsigned short* __restrict__ hbo) {
    int wave = threadIdx.x >> 5, lane = threadIdx.x & 31;
    int nt = blockIdx.y * 4 + wave;       // output tile 0..15 (256 cols)
    int r0 = blockIdx.x << 4;
    int m = lane & 15, g = lane >> 4;
    const unsigned short* xrow = xb + (size_t)(r0 + m) * 256;
    const unsigned short* hrow = hb + (size_t)(r0 + m) * 256;
    v8f gi0, gi1, gi2, gh0, gh1, gh2;
#pragma unroll
    for (int i = 0; i < 8; ++i) { gi0[i]=0.f; gi1[i]=0.f; gi2[i]=0.f; gh0[i]=0.f; gh1[i]=0.f; gh2[i]=0.f; }
    for (int kt = 0; kt < 8; ++kt) {
        v16bf ax = load_afrag(xrow + (kt << 5), g);
        v16bf ah = load_afrag(hrow + (kt << 5), g);
        v16bf b;
        b = *reinterpret_cast<const v16bf*>(Pwi + ((size_t)(kt * 48 + nt)      * 32 + lane) * 16);
        gi0 = wmma_bf16(ax, b, gi0);
        b = *reinterpret_cast<const v16bf*>(Pwi + ((size_t)(kt * 48 + 16 + nt) * 32 + lane) * 16);
        gi1 = wmma_bf16(ax, b, gi1);
        b = *reinterpret_cast<const v16bf*>(Pwi + ((size_t)(kt * 48 + 32 + nt) * 32 + lane) * 16);
        gi2 = wmma_bf16(ax, b, gi2);
        b = *reinterpret_cast<const v16bf*>(Pwh + ((size_t)(kt * 48 + nt)      * 32 + lane) * 16);
        gh0 = wmma_bf16(ah, b, gh0);
        b = *reinterpret_cast<const v16bf*>(Pwh + ((size_t)(kt * 48 + 16 + nt) * 32 + lane) * 16);
        gh1 = wmma_bf16(ah, b, gh1);
        b = *reinterpret_cast<const v16bf*>(Pwh + ((size_t)(kt * 48 + 32 + nt) * 32 + lane) * 16);
        gh2 = wmma_bf16(ah, b, gh2);
    }
    int n = (nt << 4) + m;
    float bir = bi[n], biz = bi[256 + n], binn = bi[512 + n];
    float bhr = bh[n], bhz = bh[256 + n], bhn  = bh[512 + n];
#pragma unroll
    for (int v = 0; v < 8; ++v) {
        int row = r0 + v + (g << 3);
        float r = 1.f / (1.f + expf(-(gi0[v] + bir + gh0[v] + bhr)));
        float z = 1.f / (1.f + expf(-(gi1[v] + biz + gh1[v] + bhz)));
        float nn = tanhf(gi2[v] + binn + r * (gh2[v] + bhn));
        float ho = hf[(size_t)row * 256 + n];
        float hn = (1.f - z) * nn + z * ho;
        hfo[(size_t)row * 256 + n] = hn;
        hbo[(size_t)row * 256 + n] = f2bf(hn);
    }
}

// ---------- per-step input builder: X[b,384] = [cond(256) | prev/g,logg(41) | cos,sin(80) | pad] ----------
__global__ __launch_bounds__(256) void step_pre(
    const unsigned short* __restrict__ condb, const float* __restrict__ sig,
    const int* __restrict__ period, const float* __restrict__ w0f, const float* __restrict__ cumf,
    int t, int NF_TOT, int F4, int SIGN, unsigned short* __restrict__ X) {
    int b = blockIdx.x, tid = threadIdx.x;
    int frame = t >> 2, sub = t & 3;
    __shared__ float prevs[40];
    __shared__ float red[64];
    unsigned short* Xb = X + (size_t)b * 384;
    Xb[tid] = condb[((size_t)b * F4 + frame) * 256 + tid];     // cond copy (already bf16)
    float pval = 0.f;
    if (tid < 40) {
        int pitch = period[(size_t)b * NF_TOT + 3 + frame];
        int pmax = pitch > 40 ? pitch : 40;
        int gidx = t * 40 - pmax + tid;
        pval = (gidx >= 0) ? sig[(size_t)b * SIGN + gidx] : 0.f;
        prevs[tid] = pval;
    }
    if (tid < 64) red[tid] = pval * pval;
    __syncthreads();
    for (int s = 32; s > 0; s >>= 1) {
        if (tid < s) red[tid] += red[tid + s];
        __syncthreads();
    }
    float gn = sqrtf(red[0]);
    if (tid < 40) Xb[256 + tid] = f2bf(prevs[tid] / (1e-5f + gn));
    if (tid == 40) Xb[296] = f2bf(logf(1e-5f + gn));
    if (tid < 40) {
        float ang = cumf[(size_t)b * F4 + frame] + w0f[(size_t)b * F4 + frame] * (float)(sub * 40 + tid);
        Xb[297 + tid] = f2bf(cosf(ang));
        Xb[337 + tid] = f2bf(sinf(ang));
    }
    if (tid < 7) Xb[377 + tid] = 0;     // zero pad 377..383
}

// ---------- output head: sig = tanh(h3@sdo + b) * exp(h3@gain + b) ; WMMA N=48 padded ----------
__global__ __launch_bounds__(128) void out_step(
    const unsigned short* __restrict__ h3b, const unsigned short* __restrict__ Pout,
    const float* __restrict__ sdo_b, const float* __restrict__ gain_b,
    float* __restrict__ sig, int t, int SIGN) {
    __shared__ float sv[4][16][48];
    int wave = threadIdx.x >> 5, lane = threadIdx.x & 31;
    int r0 = blockIdx.x * 64 + wave * 16;
    int m = lane & 15, g = lane >> 4;
    const unsigned short* hrow = h3b + (size_t)(r0 + m) * 256;
    v8f c[3];
#pragma unroll
    for (int tt = 0; tt < 3; ++tt)
#pragma unroll
        for (int i = 0; i < 8; ++i) c[tt][i] = 0.f;
    for (int kt = 0; kt < 8; ++kt) {
        v16bf a = load_afrag(hrow + (kt << 5), g);
#pragma unroll
        for (int tt = 0; tt < 3; ++tt) {
            v16bf b = *reinterpret_cast<const v16bf*>(Pout + ((size_t)(kt * 3 + tt) * 32 + lane) * 16);
            c[tt] = wmma_bf16(a, b, c[tt]);
        }
    }
#pragma unroll
    for (int tt = 0; tt < 3; ++tt) {
        int n = tt * 16 + m;
        float bv = (n < 40) ? sdo_b[n] : ((n == 40) ? gain_b[0] : 0.f);
#pragma unroll
        for (int v = 0; v < 8; ++v) sv[wave][v + (g << 3)][n] = c[tt][v] + bv;
    }
    __syncthreads();
    for (int e = threadIdx.x; e < 64 * 40; e += 128) {
        int rr = e / 40, k = e - rr * 40;
        float val = tanhf(sv[rr >> 4][rr & 15][k]) * expf(sv[rr >> 4][rr & 15][40]);
        sig[(size_t)(blockIdx.x * 64 + rr) * SIGN + t * 40 + k] = val;
    }
}

// ---------- phase cumsum (per batch sample) ----------
__global__ void phase_prep(const int* __restrict__ period, const float* __restrict__ prand,
                           int NF_TOT, int F4, float* __restrict__ w0f, float* __restrict__ cumf) {
    int b = blockIdx.x * blockDim.x + threadIdx.x;
    const float TWO_PI = 6.2831853071795864769f;
    float acc = TWO_PI * prand[b];
    for (int j = 0; j < F4; ++j) {
        float w0 = TWO_PI / (float)period[(size_t)b * NF_TOT + 3 + j];
        cumf[(size_t)b * F4 + j] = acc;
        w0f[(size_t)b * F4 + j] = w0;
        acc += 160.f * w0;
    }
}

// ---------- cond-net input: [features(20) | pembed[period](64) | pad(12)] bf16 ----------
__global__ void build_xfd1(const float* __restrict__ features, const int* __restrict__ period,
                           const float* __restrict__ pembed, unsigned short* __restrict__ X) {
    int row = blockIdx.x, c = threadIdx.x;
    float v = 0.f;
    if (c < 20) v = features[(size_t)row * 20 + c];
    else if (c < 84) v = pembed[(size_t)period[row] * 64 + (c - 20)];
    X[(size_t)row * 96 + c] = f2bf(v);
}

__global__ void copy_h(const float* __restrict__ h1, const float* __restrict__ h2,
                       const float* __restrict__ h3, float* __restrict__ dst, int n) {
    int i = blockIdx.x * blockDim.x + threadIdx.x;
    if (i < n) dst[i] = h1[i];
    else if (i < 2 * n) dst[i] = h2[i - n];
    else if (i < 3 * n) dst[i] = h3[i - 2 * n];
}

extern "C" void kernel_launch(void* const* d_in, const int* in_sizes, int n_in,
                              void* d_out, int out_size, void* d_ws, size_t ws_size,
                              hipStream_t stream) {
    const float* features   = (const float*)d_in[0];
    const int*   period     = (const int*)d_in[1];
    const float* phase_rand = (const float*)d_in[3];
    const float* pembed     = (const float*)d_in[4];
    const float* fd1_w = (const float*)d_in[5];  const float* fd1_b = (const float*)d_in[6];
    const float* fc1_w = (const float*)d_in[7];  const float* fc1_b = (const float*)d_in[8];
    const float* fc2_w = (const float*)d_in[9];  const float* fc2_b = (const float*)d_in[10];
    const float* fd2_w = (const float*)d_in[11]; const float* fd2_b = (const float*)d_in[12];
    const float* sd1_w = (const float*)d_in[13]; const float* sd1_b = (const float*)d_in[14];
    const float* sd2_w = (const float*)d_in[15]; const float* sd2_b = (const float*)d_in[16];
    const float* g_wi[3] = {(const float*)d_in[17], (const float*)d_in[21], (const float*)d_in[25]};
    const float* g_wh[3] = {(const float*)d_in[18], (const float*)d_in[22], (const float*)d_in[26]};
    const float* g_bi[3] = {(const float*)d_in[19], (const float*)d_in[23], (const float*)d_in[27]};
    const float* g_bh[3] = {(const float*)d_in[20], (const float*)d_in[24], (const float*)d_in[28]};
    const float* sdo_w = (const float*)d_in[29]; const float* sdo_b = (const float*)d_in[30];
    const float* gain_w = (const float*)d_in[31]; const float* gain_b = (const float*)d_in[32];

    // Derived shapes (host cannot read device scalars).
    const int B      = in_sizes[3];                   // phase_rand is [B,1]
    const int NF_TOT = in_sizes[1] / B;               // period is [B, NF_TOT]
    const int F2 = NF_TOT - 2, F4 = NF_TOT - 4;
    const int nbf  = (out_size - 3 * B * 256) / (B * 160);
    const int S    = nbf * 4;
    const int SIGN = nbf * 160;

    float* sig  = (float*)d_out;
    float* hout = sig + (size_t)B * SIGN;

    // ---- bump allocator over d_ws ----
    char* wp = (char*)d_ws;
    auto alloc = [&](size_t bytes) -> char* {
        char* r = wp;
        wp += (bytes + 255) & ~(size_t)255;
        return r;
    };
    auto au16 = [&](size_t n) { return (unsigned short*)alloc(n * 2); };
    auto af32 = [&](size_t n) { return (float*)alloc(n * 4); };

    unsigned short* Pfd1 = au16((size_t)96 * 256);
    unsigned short* Pc1  = au16((size_t)768 * 256);
    unsigned short* Pc2  = au16((size_t)768 * 256);
    unsigned short* Pfd2 = au16((size_t)256 * 256);
    unsigned short* Psd1 = au16((size_t)384 * 256);
    unsigned short* Psd2 = au16((size_t)256 * 256);
    unsigned short *Pwi[3], *Pwh[3];
    for (int gI = 0; gI < 3; ++gI) { Pwi[gI] = au16((size_t)256 * 768); Pwh[gI] = au16((size_t)256 * 768); }
    unsigned short* Pout = au16((size_t)256 * 48);
    unsigned short* Xfd1 = au16((size_t)B * NF_TOT * 96);
    unsigned short* bufA = au16((size_t)B * NF_TOT * 256);
    unsigned short* bufB = au16((size_t)B * F2 * 256);
    float* w0f  = af32((size_t)B * F4);
    float* cumf = af32((size_t)B * F4);
    unsigned short* Xstep = au16((size_t)B * 384);
    unsigned short* T1b = au16((size_t)B * 256);
    unsigned short* T2b = au16((size_t)B * 256);
    float* hf[3][2]; unsigned short* hbuf[3][2];
    for (int gI = 0; gI < 3; ++gI)
        for (int s = 0; s < 2; ++s) { hf[gI][s] = af32((size_t)B * 256); hbuf[gI][s] = au16((size_t)B * 256); }

    // ---- pack weights into WMMA B-fragment order (per launch; deterministic) ----
    pack_w<<<3 * 16, 32, 0, stream>>>(fd1_w, nullptr, 0, 84, 256, 16, Pfd1);
    pack_w<<<24 * 16, 32, 0, stream>>>(fc1_w, nullptr, 1, 768, 256, 16, Pc1);
    pack_w<<<24 * 16, 32, 0, stream>>>(fc2_w, nullptr, 1, 768, 256, 16, Pc2);
    pack_w<<<8 * 16, 32, 0, stream>>>(fd2_w, nullptr, 0, 256, 256, 16, Pfd2);
    pack_w<<<12 * 16, 32, 0, stream>>>(sd1_w, nullptr, 0, 377, 256, 16, Psd1);
    pack_w<<<8 * 16, 32, 0, stream>>>(sd2_w, nullptr, 0, 256, 256, 16, Psd2);
    for (int gI = 0; gI < 3; ++gI) {
        pack_w<<<8 * 48, 32, 0, stream>>>(g_wi[gI], nullptr, 0, 256, 768, 48, Pwi[gI]);
        pack_w<<<8 * 48, 32, 0, stream>>>(g_wh[gI], nullptr, 0, 256, 768, 48, Pwh[gI]);
    }
    pack_w<<<8 * 3, 32, 0, stream>>>(sdo_w, gain_w, 2, 256, 48, 3, Pout);

    // ---- zero initial hidden states ----
    for (int gI = 0; gI < 3; ++gI) {
        hipMemsetAsync(hf[gI][0], 0, (size_t)B * 256 * 4, stream);
        hipMemsetAsync(hbuf[gI][0], 0, (size_t)B * 256 * 2, stream);
    }

    // ---- cond net: fd1 -> conv1 -> conv2 -> fd2 (convs as overlapping-row GEMMs, K=768) ----
    build_xfd1<<<B * NF_TOT, 96, 0, stream>>>(features, period, pembed, Xfd1);
    gemm_bf16<<<dim3(B * NF_TOT / 16, 4), 128, 0, stream>>>(Xfd1, B * NF_TOT, 0L, 96,
                                                            Pfd1, 3, 16, fd1_b, 1, bufA, 256);
    gemm_bf16<<<dim3(B * F2 / 16, 4), 128, 0, stream>>>(bufA, F2, (long)NF_TOT * 256, 256,
                                                        Pc1, 24, 16, fc1_b, 1, bufB, 256);
    gemm_bf16<<<dim3(B * F4 / 16, 4), 128, 0, stream>>>(bufB, F4, (long)F2 * 256, 256,
                                                        Pc2, 24, 16, fc2_b, 1, bufA, 256);
    gemm_bf16<<<dim3(B * F4 / 16, 4), 128, 0, stream>>>(bufA, B * F4, 0L, 256,
                                                        Pfd2, 8, 16, fd2_b, 1, bufB, 256);
    const unsigned short* condb = bufB;

    phase_prep<<<B / 64, 64, 0, stream>>>(period, phase_rand, NF_TOT, F4, w0f, cumf);

    // ---- sequential scan ----
    for (int t = 0; t < S; ++t) {
        int cur = t & 1, nxt = cur ^ 1;
        step_pre<<<B, 256, 0, stream>>>(condb, sig, period, w0f, cumf, t, NF_TOT, F4, SIGN, Xstep);
        gemm_bf16<<<dim3(B / 16, 4), 128, 0, stream>>>(Xstep, B, 0L, 384, Psd1, 12, 16, sd1_b, 1, T1b, 256);
        gemm_bf16<<<dim3(B / 16, 4), 128, 0, stream>>>(T1b, B, 0L, 256, Psd2, 8, 16, sd2_b, 1, T2b, 256);
        gru_step<<<dim3(B / 16, 4), 128, 0, stream>>>(T2b, hbuf[0][cur], hf[0][cur],
                                                      Pwi[0], Pwh[0], g_bi[0], g_bh[0],
                                                      hf[0][nxt], hbuf[0][nxt]);
        gru_step<<<dim3(B / 16, 4), 128, 0, stream>>>(hbuf[0][nxt], hbuf[1][cur], hf[1][cur],
                                                      Pwi[1], Pwh[1], g_bi[1], g_bh[1],
                                                      hf[1][nxt], hbuf[1][nxt]);
        gru_step<<<dim3(B / 16, 4), 128, 0, stream>>>(hbuf[1][nxt], hbuf[2][cur], hf[2][cur],
                                                      Pwi[2], Pwh[2], g_bi[2], g_bh[2],
                                                      hf[2][nxt], hbuf[2][nxt]);
        out_step<<<B / 64, 128, 0, stream>>>(hbuf[2][nxt], Pout, sdo_b, gain_b, sig, t, SIGN);
    }

    // ---- final hidden states to d_out tail (S even -> final state in buffer S&1) ----
    int fin = S & 1;
    copy_h<<<(3 * B * 256 + 255) / 256, 256, 0, stream>>>(hf[0][fin], hf[1][fin], hf[2][fin],
                                                          hout, B * 256);
}
